// WrappedLoRALinear_79791902425196
// MI455X (gfx1250) — compile-verified, matrix operand order checked by reading, not run on hardware
//
#include <hip/hip_runtime.h>
#include <hip/hip_bf16.h>
#include <stdint.h>

// Problem constants (match reference).
#define BATCH 4
#define SEQ   2048
#define DIN   4096
#define DOUT  4096
#define RANK  16
#define AUXD  64
#define AUXU  64
// ALPHA / RANK = 32/16 = 2.0, folded into the 'up' factor at build time.

typedef __attribute__((ext_vector_type(16))) __bf16 v16bf;
typedef __attribute__((ext_vector_type(8)))  float  v8f;

union FragBF {
    v16bf v;
    uint4 q[2];
};

// ---- CDNA5 async global->LDS copy (ASYNCcnt path), guarded. -------------
// Probe round 2 confirmed the builtin exists with prototype:
//   void (v4i AS1* gsrc, v4i AS3* lds_dst, imm int offset, imm int cpol)
// where v4i = int __attribute__((vector_size(16))).
#if defined(__HIP_DEVICE_COMPILE__) && __has_builtin(__builtin_amdgcn_global_load_async_to_lds_b128)
#define HAVE_ASYNC_LDS 1
typedef int v4i_raw __attribute__((vector_size(16)));
typedef __attribute__((address_space(1))) v4i_raw* g_v4i_ptr;
typedef __attribute__((address_space(3))) v4i_raw* l_v4i_ptr;
#define ASYNC_CP128(lds, glb)                                              \
    __builtin_amdgcn_global_load_async_to_lds_b128(                        \
        (g_v4i_ptr)(glb), (l_v4i_ptr)(lds), 0, 0)
#if __has_builtin(__builtin_amdgcn_s_wait_asynccnt)
#define WAIT_ASYNC0() __builtin_amdgcn_s_wait_asynccnt(0)
#else
#define WAIT_ASYNC0() asm volatile("s_wait_asynccnt 0" ::: "memory")
#endif
#else
#define HAVE_ASYNC_LDS 0
#endif

__device__ __forceinline__ unsigned short f2bf(float f) {
    unsigned u = __float_as_uint(f);
    unsigned r = u + 0x7FFFu + ((u >> 16) & 1u);   // round-to-nearest-even
    return (unsigned short)(r >> 16);
}

// ---------------------------------------------------------------------------
// fp32 -> bf16 conversion, 8 elements / thread, 16B store.
// ---------------------------------------------------------------------------
__global__ void cvt_bf16_kernel(const float* __restrict__ in,
                                unsigned short* __restrict__ out, long n) {
    long i = ((long)blockIdx.x * blockDim.x + threadIdx.x) * 8;
    if (i >= n) return;
    float4 v0 = *(const float4*)&in[i];
    float4 v1 = *(const float4*)&in[i + 4];
    union { unsigned short s[8]; uint4 q; } o;
    o.s[0] = f2bf(v0.x); o.s[1] = f2bf(v0.y); o.s[2] = f2bf(v0.z); o.s[3] = f2bf(v0.w);
    o.s[4] = f2bf(v1.x); o.s[5] = f2bf(v1.y); o.s[6] = f2bf(v1.z); o.s[7] = f2bf(v1.w);
    *(uint4*)&out[i] = o.q;
}

// ---------------------------------------------------------------------------
// down[b, r, i] = sum_a down_aux[i, a] * hyper_down[b, a, r]   (fp32)
// ---------------------------------------------------------------------------
__global__ void build_down_kernel(const float* __restrict__ down_aux,
                                  const float* __restrict__ hyper_down,
                                  float* __restrict__ down) {
    int idx = blockIdx.x * blockDim.x + threadIdx.x;    // b*2^16 + r*2^12 + i
    if (idx >= BATCH * RANK * DIN) return;
    int i = idx & (DIN - 1);
    int r = (idx >> 12) & (RANK - 1);
    int b = idx >> 16;
    float acc = 0.f;
#pragma unroll 8
    for (int a = 0; a < AUXD; ++a)
        acc += down_aux[(size_t)i * AUXD + a] *
               hyper_down[((size_t)b * AUXD + a) * RANK + r];
    down[idx] = acc;
}

// ---------------------------------------------------------------------------
// up[b, o, r] = (ALPHA/RANK) * sum_a up_aux[a, o] * hyper_up[b, r, a]  (fp32)
// ---------------------------------------------------------------------------
__global__ void build_up_kernel(const float* __restrict__ up_aux,
                                const float* __restrict__ hyper_up,
                                float* __restrict__ up) {
    int idx = blockIdx.x * blockDim.x + threadIdx.x;    // b*2^16 + o*2^4 + r
    if (idx >= BATCH * DOUT * RANK) return;
    int r = idx & (RANK - 1);
    int o = (idx >> 4) & (DOUT - 1);
    int b = idx >> 16;
    float acc = 0.f;
#pragma unroll 8
    for (int a = 0; a < AUXU; ++a)
        acc += up_aux[(size_t)a * DOUT + o] *
               hyper_up[((size_t)b * RANK + r) * AUXU + a];
    up[idx] = acc * 2.0f;                               // fold alpha/rank
}

// ---------------------------------------------------------------------------
// low_rank[b, s, r] = sum_i x[b, s, i] * down[b, r, i]   (fp32, LDS staged)
// ---------------------------------------------------------------------------
__global__ void lowrank_kernel(const float* __restrict__ x,
                               const float* __restrict__ down,
                               float* __restrict__ lr) {
    __shared__ float xs[16][68];
    __shared__ float ds[16][68];
    int b  = blockIdx.x >> 7;                 // SEQ/16 = 128 blocks per batch
    int s0 = (blockIdx.x & 127) * 16;
    int t  = threadIdx.x;
    int row = t & 15;
    int c4  = (t >> 4) * 4;
    int sl  = t >> 4;
    int r   = t & 15;
    float acc = 0.f;
    for (int k0 = 0; k0 < DIN; k0 += 64) {
        __syncthreads();
        float4 xv = *(const float4*)&x[(((size_t)b * SEQ) + s0 + row) * DIN + k0 + c4];
        float4 dv = *(const float4*)&down[(((size_t)b * RANK) + row) * DIN + k0 + c4];
        *(float4*)&xs[row][c4] = xv;
        *(float4*)&ds[row][c4] = dv;
        __syncthreads();
#pragma unroll
        for (int kk = 0; kk < 64; ++kk)
            acc += xs[sl][kk] * ds[r][kk];
    }
    lr[(((size_t)b * SEQ) + s0 + sl) * RANK + r] = acc;
}

// ---------------------------------------------------------------------------
// Main fused GEMM:
//   out[b, s, o] = sum_i x[b,s,i] W[o,i] + bias[o] + sum_r lr[b,s,r]*up[b,o,r]
// bf16 WMMA, 128x128 block tile, BK=64, 8 waves (wave32), each wave a 64x32
// region (4x2 tiles). Double-buffered LDS filled by async global->LDS copies
// (ASYNCcnt) when available; rows padded to 72 bf16 for conflict-free b128
// fragment reads (36-bank stride).
// ---------------------------------------------------------------------------
#define LROW 72                      // LDS row stride in bf16 units
#define BK   64

__launch_bounds__(256, 2)
__global__ void lora_gemm_kernel(const unsigned short* __restrict__ xbf,
                                 const unsigned short* __restrict__ wbf,
                                 const float* __restrict__ bias,
                                 const float* __restrict__ upw,
                                 const float* __restrict__ lr,
                                 float* __restrict__ out) {
    __shared__ __align__(16) unsigned short As[2][128 * LROW];
    __shared__ __align__(16) unsigned short Bs[2][128 * LROW];
    __shared__ __align__(16) float lr_s[128][16];
    __shared__ __align__(16) float up_s[128][16];
    __shared__ float bias_s[128];

    const int tid  = threadIdx.x;
    const int lane = tid & 31;
    const int wave = tid >> 5;

    const int bidx = blockIdx.x;                 // 2048 = 4 * 16 * 32
    const int b    = bidx >> 9;
    const int mblk = (bidx >> 5) & 15;
    const int nblk = bidx & 31;
    const int s0   = mblk * 128;
    const int n0   = nblk * 128;

    const int wm = (wave & 1) * 64;              // wave M origin (0 / 64)
    const int wn = (wave >> 1) * 32;             // wave N origin (0/32/64/96)

    // Global-tile mapping: 256 threads move 128x64 bf16 as 4 x 16B each.
    const int rowG  = tid >> 1;                  // 0..127
    const int cbase = (tid & 1) * 32;            // bf16 units
    const size_t xbase = ((size_t)b * SEQ + s0) * DIN;
    const size_t wbase = (size_t)n0 * DIN;

    // Stage epilogue data (independent of K loop).
    {
        int m  = tid >> 1;
        int r0 = (tid & 1) * 8;
        const float4* plr = (const float4*)&lr[(((size_t)b * SEQ) + s0 + m) * RANK + r0];
        *(float4*)&lr_s[m][r0]     = plr[0];
        *(float4*)&lr_s[m][r0 + 4] = plr[1];
        const float4* pup = (const float4*)&upw[(((size_t)b * DOUT) + n0 + m) * RANK + r0];
        *(float4*)&up_s[m][r0]     = pup[0];
        *(float4*)&up_s[m][r0 + 4] = pup[1];
        if (tid < 128) bias_s[tid] = bias[n0 + tid];
    }

    v8f acc[4][2];
    v8f zero = {0.f, 0.f, 0.f, 0.f, 0.f, 0.f, 0.f, 0.f};
#pragma unroll
    for (int i = 0; i < 4; ++i)
#pragma unroll
        for (int j = 0; j < 2; ++j) acc[i][j] = zero;

    // Fragment addressing (CDNA5 16-bit A/B layout): lane holds matrix row
    // (lane & 15); K-half selected by lane>>4: 16B chunk at +0/+16, second
    // chunk at +32/+48 within each 32-element K group.
    const int fr = lane & 15;
    const int fo = (lane >> 4) * 8;              // ushort offset (16 B)

#if HAVE_ASYNC_LDS
    // ---- async pipeline: fill buffer 0 for k-tile 0 ----
    {
#pragma unroll
        for (int c = 0; c < 4; ++c) {
            const int col = cbase + c * 8;
            ASYNC_CP128(&As[0][rowG * LROW + col],
                        &xbf[xbase + (size_t)rowG * DIN + col]);
            ASYNC_CP128(&Bs[0][rowG * LROW + col],
                        &wbf[wbase + (size_t)rowG * DIN + col]);
        }
        WAIT_ASYNC0();
    }
    __syncthreads();

    for (int kt = 0; kt < DIN / BK; ++kt) {
        const int buf = kt & 1;
        const bool pre = (kt < DIN / BK - 1);
        if (pre) {
            const int k0n = (kt + 1) * BK;
#pragma unroll
            for (int c = 0; c < 4; ++c) {
                const int col = cbase + c * 8;
                ASYNC_CP128(&As[buf ^ 1][rowG * LROW + col],
                            &xbf[xbase + (size_t)rowG * DIN + k0n + col]);
                ASYNC_CP128(&Bs[buf ^ 1][rowG * LROW + col],
                            &wbf[wbase + (size_t)rowG * DIN + k0n + col]);
            }
        }
#pragma unroll
        for (int ks = 0; ks < 2; ++ks) {
            FragBF af[4], bg[2];
#pragma unroll
            for (int i = 0; i < 4; ++i) {
                const unsigned short* p =
                    &As[buf][(wm + 16 * i + fr) * LROW + ks * 32 + fo];
                af[i].q[0] = *(const uint4*)p;
                af[i].q[1] = *(const uint4*)(p + 16);
            }
#pragma unroll
            for (int j = 0; j < 2; ++j) {
                const unsigned short* p =
                    &Bs[buf][(wn + 16 * j + fr) * LROW + ks * 32 + fo];
                bg[j].q[0] = *(const uint4*)p;
                bg[j].q[1] = *(const uint4*)(p + 16);
            }
#pragma unroll
            for (int i = 0; i < 4; ++i)
#pragma unroll
                for (int j = 0; j < 2; ++j)
                    acc[i][j] = __builtin_amdgcn_wmma_f32_16x16x32_bf16(
                        false, af[i].v, false, bg[j].v,
                        (short)0, acc[i][j], false, false);
        }
        if (pre) {
            WAIT_ASYNC0();           // my copies into buf^1 have landed
            __syncthreads();         // everyone's copies have landed
        }
    }
#else
    // ---- fallback: synchronous single-buffer pipeline ----
    for (int kt = 0; kt < DIN / BK; ++kt) {
        const int k0 = kt * BK;
        __syncthreads();
#pragma unroll
        for (int c = 0; c < 4; ++c) {
            const int col = cbase + c * 8;
            *(uint4*)&As[0][rowG * LROW + col] =
                *(const uint4*)&xbf[xbase + (size_t)rowG * DIN + k0 + col];
            *(uint4*)&Bs[0][rowG * LROW + col] =
                *(const uint4*)&wbf[wbase + (size_t)rowG * DIN + k0 + col];
        }
        __syncthreads();
#pragma unroll
        for (int ks = 0; ks < 2; ++ks) {
            FragBF af[4], bg[2];
#pragma unroll
            for (int i = 0; i < 4; ++i) {
                const unsigned short* p =
                    &As[0][(wm + 16 * i + fr) * LROW + ks * 32 + fo];
                af[i].q[0] = *(const uint4*)p;
                af[i].q[1] = *(const uint4*)(p + 16);
            }
#pragma unroll
            for (int j = 0; j < 2; ++j) {
                const unsigned short* p =
                    &Bs[0][(wn + 16 * j + fr) * LROW + ks * 32 + fo];
                bg[j].q[0] = *(const uint4*)p;
                bg[j].q[1] = *(const uint4*)(p + 16);
            }
#pragma unroll
            for (int i = 0; i < 4; ++i)
#pragma unroll
                for (int j = 0; j < 2; ++j)
                    acc[i][j] = __builtin_amdgcn_wmma_f32_16x16x32_bf16(
                        false, af[i].v, false, bg[j].v,
                        (short)0, acc[i][j], false, false);
        }
    }
#endif

    // Epilogue: bias + rank-16 low-rank delta, fp32 store.
    // C/D layout: lanes 0-15 -> N=lane, M=e; lanes 16-31 -> N=lane-16, M=8+e.
    const int nl  = lane & 15;
    const int mhi = (lane >> 4) << 3;
#pragma unroll
    for (int i = 0; i < 4; ++i) {
#pragma unroll
        for (int j = 0; j < 2; ++j) {
            const int nloc = wn + 16 * j + nl;
#pragma unroll
            for (int e = 0; e < 8; ++e) {
                const int mloc = wm + 16 * i + mhi + e;
                float d = 0.f;
#pragma unroll
                for (int r = 0; r < RANK; ++r)
                    d += lr_s[mloc][r] * up_s[nloc][r];
                out[((size_t)b * SEQ + s0 + mloc) * DOUT + n0 + nloc] =
                    acc[i][j][e] + bias_s[nloc] + d;
            }
        }
    }
}

// ---------------------------------------------------------------------------
// Host-side launcher. Workspace layout (bytes), total ~99 MB:
//   [0,            67108864)  x_bf16   (4*2048*4096 bf16)
//   [67108864,    100663296)  W_bf16   (4096*4096 bf16)
//   [100663296,   101711872)  down     fp32 [B][16][4096]
//   [101711872,   102760448)  up*scale fp32 [B][4096][16]
//   [102760448,   103284736)  low_rank fp32 [B][2048][16]
// ---------------------------------------------------------------------------
extern "C" void kernel_launch(void* const* d_in, const int* in_sizes, int n_in,
                              void* d_out, int out_size, void* d_ws, size_t ws_size,
                              hipStream_t stream) {
    const float* x          = (const float*)d_in[0];
    const float* W          = (const float*)d_in[1];
    const float* bias       = (const float*)d_in[2];
    const float* down_aux   = (const float*)d_in[3];
    const float* up_aux     = (const float*)d_in[4];
    const float* hyper_down = (const float*)d_in[5];
    const float* hyper_up   = (const float*)d_in[6];
    float* out = (float*)d_out;

    char* ws = (char*)d_ws;
    unsigned short* xbf  = (unsigned short*)(ws);
    unsigned short* wbf  = (unsigned short*)(ws + 67108864);
    float*          down = (float*)(ws + 100663296);
    float*          upw  = (float*)(ws + 101711872);
    float*          lrw  = (float*)(ws + 102760448);

    const long nx = (long)BATCH * SEQ * DIN;     // 33554432
    const long nw = (long)DOUT * DIN;            // 16777216

    cvt_bf16_kernel<<<(unsigned)(nx / 8 / 256), 256, 0, stream>>>(x, xbf, nx);
    cvt_bf16_kernel<<<(unsigned)(nw / 8 / 256), 256, 0, stream>>>(W, wbf, nw);
    build_down_kernel<<<(BATCH * RANK * DIN) / 256, 256, 0, stream>>>(down_aux, hyper_down, down);
    build_up_kernel<<<(BATCH * DOUT * RANK) / 256, 256, 0, stream>>>(up_aux, hyper_up, upw);
    lowrank_kernel<<<BATCH * (SEQ / 16), 256, 0, stream>>>(x, down, lrw);
    lora_gemm_kernel<<<BATCH * (SEQ / 128) * (DOUT / 128), 256, 0, stream>>>(
        xbf, wbf, bias, upw, lrw, out);
}
